// ResnetCnnsovnetDynamicRouting_867583394572
// MI455X (gfx1250) — matrix operand order repaired and checked by reading, not compiled, not run on hardware
//
#include <hip/hip_runtime.h>
#include <hip/hip_bf16.h>
#include <math.h>

typedef __attribute__((ext_vector_type(16))) _Float16 v16h;
typedef __attribute__((ext_vector_type(8)))  _Float16 v8h;
typedef __attribute__((ext_vector_type(4)))  _Float16 v4h;
typedef __attribute__((ext_vector_type(8)))  float    v8f;

// ---------------------------------------------------------------------------
// Implicit-GEMM convolution with WMMA (f16 inputs, f32 accumulate).
// x: (Nb, Cin, H, W) f32, w: (Cout, Cin, KH, KW) f32, out: (Nb, Cout, Hp, Wp)
// Block: 32 output pixels x Cout channels. Each wave owns one 16-channel
// column group and two 16-pixel M-subtiles. K-step = 64 -> 4 WMMAs per
// staging phase per wave. LDS rows are 144B so all fragment chunks are
// 16B-aligned ds_load_b128.
// ---------------------------------------------------------------------------
__global__ void conv_wmma(const float* __restrict__ x, const float* __restrict__ w,
                          const float* __restrict__ bias, float* __restrict__ out,
                          int Cin, int H, int W, int Cout, int KH, int KW,
                          int stride, int Hp, int Wp, int act) {
  const int nb    = blockIdx.x;
  const int mtile = blockIdx.y;           // 32 pixels per tile
  const int P     = Hp * Wp;
  const int Ktot  = Cin * KH * KW;

  __shared__ _Float16 As[32][72];         // 32 pixels x 64 K (+pad), rows 144B
  __shared__ _Float16 Bt[128][72];        // Cout x 64 K transposed (+pad)

  const int tid  = threadIdx.x;
  const int lane = tid & 31;
  const int wv   = tid >> 5;
  const int n0   = wv * 16;
  const int row  = lane & 15;
  const int hi   = lane >> 4;

  v8f acc0 = {};
  v8f acc1 = {};

  for (int k0 = 0; k0 < Ktot; k0 += 64) {
    // ---- stage A (im2col gather, 32 x 64) ----
    for (int idx = tid; idx < 2048; idx += blockDim.x) {
      int m  = idx >> 6;
      int kk = idx & 63;
      int k  = k0 + kk;
      int p  = mtile * 32 + m;
      float v = 0.f;
      if (k < Ktot && p < P) {
        int ci  = k / (KH * KW);
        int rem = k % (KH * KW);
        int ky  = rem / KW, kx = rem % KW;
        int oy  = p / Wp,   ox = p % Wp;
        int iy  = oy * stride + ky;
        int ix  = ox * stride + kx;
        v = x[((size_t)(nb * Cin + ci) * H + iy) * W + ix];
      }
      As[m][kk] = (_Float16)v;
    }
    // ---- stage B transposed: Bt[n][kk], one ds_store_b64 per 4-k run ----
    for (int idx = tid; idx < 16 * Cout; idx += blockDim.x) {
      int n  = idx >> 4;
      int kq = (idx & 15) * 4;
      v4h hv;
#pragma unroll
      for (int j = 0; j < 4; ++j) {
        int k = k0 + kq + j;
        hv[j] = (_Float16)((k < Ktot) ? w[(size_t)n * Ktot + k] : 0.f);
      }
      *(v4h*)&Bt[n][kq] = hv;
    }
    __syncthreads();

    // ---- 2 K-halves x 2 M-subtiles = 4 WMMAs per staging phase ----
#pragma unroll
    for (int kh = 0; kh < 2; ++kh) {
      const int kb = kh * 32;
      v8h a0lo = *(const v8h*)&As[row][kb + hi * 8];
      v8h a0hi = *(const v8h*)&As[row][kb + 16 + hi * 8];
      v8h a1lo = *(const v8h*)&As[16 + row][kb + hi * 8];
      v8h a1hi = *(const v8h*)&As[16 + row][kb + 16 + hi * 8];
      v8h blo  = *(const v8h*)&Bt[n0 + row][kb + hi * 8];
      v8h bhi  = *(const v8h*)&Bt[n0 + row][kb + 16 + hi * 8];
      v16h a0 = __builtin_shufflevector(a0lo, a0hi, 0, 1, 2, 3, 4, 5, 6, 7,
                                        8, 9, 10, 11, 12, 13, 14, 15);
      v16h a1 = __builtin_shufflevector(a1lo, a1hi, 0, 1, 2, 3, 4, 5, 6, 7,
                                        8, 9, 10, 11, 12, 13, 14, 15);
      v16h bf = __builtin_shufflevector(blo, bhi, 0, 1, 2, 3, 4, 5, 6, 7,
                                        8, 9, 10, 11, 12, 13, 14, 15);
      acc0 = __builtin_amdgcn_wmma_f32_16x16x32_f16(false, a0, false, bf,
                                                    (short)0, acc0, false, false);
      acc1 = __builtin_amdgcn_wmma_f32_16x16x32_f16(false, a1, false, bf,
                                                    (short)0, acc1, false, false);
    }
    __syncthreads();
  }

  // ---- store C (lane = N col; VGPR r -> M = r + 8*hi); p contiguous in r ----
  const int col = n0 + row;
  float bv = (col < Cout) ? bias[col] : 0.f;
#pragma unroll
  for (int r = 0; r < 8; ++r) {
    int p = mtile * 32 + r + hi * 8;
    if (p < P && col < Cout) {
      float v = acc0[r] + bv;
      if (act == 1) v = fmaxf(v, 0.f);
      out[((size_t)(nb * Cout + col)) * P + p] = v;
    }
  }
#pragma unroll
  for (int r = 0; r < 8; ++r) {
    int p = mtile * 32 + 16 + r + hi * 8;
    if (p < P && col < Cout) {
      float v = acc1[r] + bv;
      if (act == 1) v = fmaxf(v, 0.f);
      out[((size_t)(nb * Cout + col)) * P + p] = v;
    }
  }
}

// ---------------------------------------------------------------------------
// Generic GEMM with WMMA: out = act(A(MxK) @ W(KxN) + bias). Block 256,
// grid (ceil(M/32), N/128). K-step 64; float4 global staging for A,
// per-thread 4-k runs + ds_store_b64 for B. K and N are multiples of 4.
// ---------------------------------------------------------------------------
__global__ void gemm_wmma(const float* __restrict__ A, const float* __restrict__ Wm,
                          const float* __restrict__ bias, float* __restrict__ out,
                          int M, int K, int N, int act) {
  const int mtile = blockIdx.x;           // 32 rows
  const int ngrp  = blockIdx.y;           // 128 cols

  __shared__ _Float16 As[32][72];
  __shared__ _Float16 Bt[128][72];

  const int tid  = threadIdx.x;
  const int lane = tid & 31;
  const int wv   = tid >> 5;
  const int n0   = wv * 16;
  const int row  = lane & 15;
  const int hi   = lane >> 4;

  v8f acc0 = {};
  v8f acc1 = {};

  for (int k0 = 0; k0 < K; k0 += 64) {
    // A: 32 rows x 16 quads; float4 global load + ds_store_b64
    for (int idx = tid; idx < 512; idx += 256) {
      int m  = idx >> 4;
      int kq = (idx & 15) * 4;
      int mg = mtile * 32 + m;
      int k  = k0 + kq;
      v4h hv = {};
      if (mg < M && k + 3 < K) {
        float4 f = *(const float4*)&A[(size_t)mg * K + k];
        hv[0] = (_Float16)f.x; hv[1] = (_Float16)f.y;
        hv[2] = (_Float16)f.z; hv[3] = (_Float16)f.w;
      } else if (mg < M) {
#pragma unroll
        for (int j = 0; j < 4; ++j)
          hv[j] = (_Float16)((k + j < K) ? A[(size_t)mg * K + k + j] : 0.f);
      }
      *(v4h*)&As[m][kq] = hv;
    }
    // B: 128 cols x 16 quads of k; 4 strided loads + one ds_store_b64
    for (int idx = tid; idx < 2048; idx += 256) {
      int n  = idx >> 4;
      int kq = (idx & 15) * 4;
      int col = ngrp * 128 + n;
      v4h hv;
#pragma unroll
      for (int j = 0; j < 4; ++j) {
        int k = k0 + kq + j;
        hv[j] = (_Float16)((k < K && col < N) ? Wm[(size_t)k * N + col] : 0.f);
      }
      *(v4h*)&Bt[n][kq] = hv;
    }
    __syncthreads();

#pragma unroll
    for (int kh = 0; kh < 2; ++kh) {
      const int kb = kh * 32;
      v8h a0lo = *(const v8h*)&As[row][kb + hi * 8];
      v8h a0hi = *(const v8h*)&As[row][kb + 16 + hi * 8];
      v8h a1lo = *(const v8h*)&As[16 + row][kb + hi * 8];
      v8h a1hi = *(const v8h*)&As[16 + row][kb + 16 + hi * 8];
      v8h blo  = *(const v8h*)&Bt[n0 + row][kb + hi * 8];
      v8h bhi  = *(const v8h*)&Bt[n0 + row][kb + 16 + hi * 8];
      v16h a0 = __builtin_shufflevector(a0lo, a0hi, 0, 1, 2, 3, 4, 5, 6, 7,
                                        8, 9, 10, 11, 12, 13, 14, 15);
      v16h a1 = __builtin_shufflevector(a1lo, a1hi, 0, 1, 2, 3, 4, 5, 6, 7,
                                        8, 9, 10, 11, 12, 13, 14, 15);
      v16h bf = __builtin_shufflevector(blo, bhi, 0, 1, 2, 3, 4, 5, 6, 7,
                                        8, 9, 10, 11, 12, 13, 14, 15);
      acc0 = __builtin_amdgcn_wmma_f32_16x16x32_f16(false, a0, false, bf,
                                                    (short)0, acc0, false, false);
      acc1 = __builtin_amdgcn_wmma_f32_16x16x32_f16(false, a1, false, bf,
                                                    (short)0, acc1, false, false);
    }
    __syncthreads();
  }

  const int col = ngrp * 128 + n0 + row;
  float bv = (col < N) ? bias[col] : 0.f;
#pragma unroll
  for (int r = 0; r < 8; ++r) {
    int m = mtile * 32 + r + hi * 8;
    if (m < M && col < N) {
      float v = acc0[r] + bv;
      if (act == 1)      v = fmaxf(v, 0.f);
      else if (act == 2) v = 1.f / (1.f + expf(-v));
      out[(size_t)m * N + col] = v;
    }
  }
#pragma unroll
  for (int r = 0; r < 8; ++r) {
    int m = mtile * 32 + 16 + r + hi * 8;
    if (m < M && col < N) {
      float v = acc1[r] + bv;
      if (act == 1)      v = fmaxf(v, 0.f);
      else if (act == 2) v = 1.f / (1.f + expf(-v));
      out[(size_t)m * N + col] = v;
    }
  }
}

// ---------------------------------------------------------------------------
// LayerNorm over (C,H,W) per sample, in place. grid = B, block = 256.
// ---------------------------------------------------------------------------
__global__ void layernorm_kernel(float* __restrict__ x, const float* __restrict__ g,
                                 const float* __restrict__ b, int n) {
  float* xb = x + (size_t)blockIdx.x * n;
  __shared__ float s1[256], s2[256];
  __shared__ float mu, rstd;
  float a = 0.f, q = 0.f;
  for (int i = threadIdx.x; i < n; i += 256) { float v = xb[i]; a += v; q += v * v; }
  s1[threadIdx.x] = a; s2[threadIdx.x] = q;
  __syncthreads();
  for (int off = 128; off > 0; off >>= 1) {
    if ((int)threadIdx.x < off) {
      s1[threadIdx.x] += s1[threadIdx.x + off];
      s2[threadIdx.x] += s2[threadIdx.x + off];
    }
    __syncthreads();
  }
  if (threadIdx.x == 0) {
    float m = s1[0] / n;
    mu = m;
    rstd = rsqrtf(s2[0] / n - m * m + 1e-5f);
  }
  __syncthreads();
  for (int i = threadIdx.x; i < n; i += 256)
    xb[i] = (xb[i] - mu) * rstd * g[i] + b[i];
}

// squash over the 16-dim capsule axis, elements strided by P. In place.
__global__ void squash_primary(float* __restrict__ x, int Ncap, int P) {
  int id = blockIdx.x * blockDim.x + threadIdx.x;
  if (id >= Ncap * P) return;
  int nc = id / P, p = id % P;
  size_t base = ((size_t)nc * 16) * P + p;
  float sq = 0.f;
#pragma unroll
  for (int d = 0; d < 16; ++d) { float v = x[base + (size_t)d * P]; sq += v * v; }
  float f = (sq / (1.f + sq)) * rsqrtf(sq + 1e-8f);
#pragma unroll
  for (int d = 0; d < 16; ++d) x[base + (size_t)d * P] *= f;
}

// ---------------------------------------------------------------------------
// Dynamic routing: one wave per spatial position, votes cached in LDS.
// u: (B, IC=8, OC, 16, P) -> vout: (B, OC, 16, P). Entropy sum accumulated.
// ---------------------------------------------------------------------------
template <int OC>
__global__ void routing_kernel(const float* __restrict__ u, float* __restrict__ vout,
                               float* __restrict__ ent_accum, int B, int P) {
  constexpr int IC = 8, D = 16, WPB = 8;
  __shared__ float uS[WPB][IC * OC * D];
  __shared__ float lgS[WPB][IC * OC];
  __shared__ float cS[WPB][IC * OC];
  __shared__ float sS[WPB][OC * D];
  __shared__ float fS[WPB][OC];
  __shared__ float red[WPB][32];

  const int tid  = threadIdx.x;
  const int lane = tid & 31;
  const int w    = tid >> 5;
  const int pos  = blockIdx.x * WPB + w;
  const bool valid = pos < B * P;
  const int b = valid ? (pos / P) : 0;
  const int p = valid ? (pos % P) : 0;

  for (int j = lane; j < IC * OC * D; j += 32) {
    int ic = j / (OC * D);
    int r  = j % (OC * D);
    uS[w][j] = valid ? u[((size_t)(b * IC + ic) * (OC * 16) + r) * P + p] : 0.f;
  }
  for (int j = lane; j < IC * OC; j += 32) lgS[w][j] = 0.f;
  __syncthreads();

  for (int it = 0; it < 3; ++it) {
    if (lane < IC) {                       // softmax over out_caps per in_cap
      float mx = -1e30f;
      for (int oc = 0; oc < OC; ++oc) mx = fmaxf(mx, lgS[w][lane * OC + oc]);
      float e[OC], sum = 0.f;
      for (int oc = 0; oc < OC; ++oc) { e[oc] = expf(lgS[w][lane * OC + oc] - mx); sum += e[oc]; }
      float inv = 1.f / sum;
      for (int oc = 0; oc < OC; ++oc) cS[w][lane * OC + oc] = e[oc] * inv;
    }
    __syncthreads();
    for (int j = lane; j < OC * D; j += 32) {   // s = sum_ic c*u
      int oc = j / D, d = j % D;
      float s = 0.f;
      for (int ic = 0; ic < IC; ++ic)
        s += cS[w][ic * OC + oc] * uS[w][(ic * OC + oc) * D + d];
      sS[w][j] = s;
    }
    __syncthreads();
    if (lane < OC) {                       // squash factor per out_cap
      float sq = 0.f;
      for (int d = 0; d < D; ++d) { float t = sS[w][lane * D + d]; sq += t * t; }
      fS[w][lane] = (sq / (1.f + sq)) * rsqrtf(sq + 1e-8f);
    }
    __syncthreads();
    for (int j = lane; j < OC * D; j += 32) sS[w][j] *= fS[w][j / D];
    __syncthreads();
    if (it < 2) {                          // agreement update
      for (int j = lane; j < IC * OC; j += 32) {
        int oc = j % OC;
        float dot = 0.f;
        for (int d = 0; d < D; ++d) dot += uS[w][j * D + d] * sS[w][oc * D + d];
        lgS[w][j] += dot;
      }
      __syncthreads();
    }
  }

  // entropy contribution: sum_{ic,oc} c*log(c)
  float ps = 0.f;
  for (int j = lane; j < IC * OC; j += 32) {
    float cv = cS[w][j];
    ps += cv * logf(cv + 1e-12f);
  }
  red[w][lane] = ps;
  __syncthreads();
  if (lane == 0 && valid) {
    float tot = 0.f;
    for (int i = 0; i < 32; ++i) tot += red[w][i];
    atomicAdd(ent_accum, tot);
  }
  for (int j = lane; j < OC * D; j += 32) {
    if (valid) {
      int oc = j / D, d = j % D;
      vout[((size_t)(b * OC + oc) * D + d) * P + p] = sS[w][j];
    }
  }
}

__global__ void init_misc(float* misc) {
  if (threadIdx.x < 16) misc[threadIdx.x] = 0.f;
}

__global__ void sel_kernel(const float* __restrict__ cc, const int* __restrict__ target,
                           float* __restrict__ sel) {
  int i = blockIdx.x * blockDim.x + threadIdx.x;
  if (i >= 64 * 16) return;
  int b = i >> 4, d = i & 15;
  sel[i] = cc[(b * 2 + target[b]) * 16 + d];
}

__global__ void class_pred(const float* __restrict__ cc, float* __restrict__ out) {
  int b = threadIdx.x;
  if (b >= 64) return;
  float s0 = 0.f, s1 = 0.f;
  for (int d = 0; d < 16; ++d) {
    float a = cc[(b * 2) * 16 + d];     s0 += a * a;
    float c = cc[(b * 2 + 1) * 16 + d]; s1 += c * c;
  }
  float n0 = sqrtf(s0 + 1e-8f), n1 = sqrtf(s1 + 1e-8f);
  float mx = fmaxf(n0, n1);
  float e0 = expf(n0 - mx), e1 = expf(n1 - mx);
  float inv = 1.f / (e0 + e1);
  out[b * 2] = e0 * inv;
  out[b * 2 + 1] = e1 * inv;
}

__global__ void ent_finalize(const float* __restrict__ misc, float* __restrict__ out) {
  if (threadIdx.x == 0) {
    out[0] = -(misc[0] / (64.f * 8.f * 784.f) + misc[1] / (64.f * 8.f * 169.f) +
               misc[2] / (64.f * 8.f * 81.f)  + misc[3] / (64.f * 8.f * 16.f) +
               misc[4] / (64.f * 8.f * 1.f));
  }
}

// ---------------------------------------------------------------------------
extern "C" void kernel_launch(void* const* d_in, const int* in_sizes, int n_in,
                              void* d_out, int out_size, void* d_ws, size_t ws_size,
                              hipStream_t stream) {
  const float* x       = (const float*)d_in[0];
  const int*   target  = (const int*)  d_in[1];
  const float* conv1_w = (const float*)d_in[2];
  const float* conv1_b = (const float*)d_in[3];
  const float* ln1_w   = (const float*)d_in[4];
  const float* ln1_b   = (const float*)d_in[5];
  const float* conv2_w = (const float*)d_in[6];
  const float* conv2_b = (const float*)d_in[7];
  const float* ln2_w   = (const float*)d_in[8];
  const float* ln2_b   = (const float*)d_in[9];
  const float* pc_w    = (const float*)d_in[10];
  const float* pc_b    = (const float*)d_in[11];
  const float* w1 = (const float*)d_in[12]; const float* b1 = (const float*)d_in[13];
  const float* w2 = (const float*)d_in[14]; const float* b2 = (const float*)d_in[15];
  const float* w3 = (const float*)d_in[16]; const float* b3 = (const float*)d_in[17];
  const float* w4 = (const float*)d_in[18]; const float* b4 = (const float*)d_in[19];
  const float* w5 = (const float*)d_in[20]; const float* b5 = (const float*)d_in[21];
  const float* fc1_w = (const float*)d_in[22]; const float* fc1_b = (const float*)d_in[23];
  const float* fc2_w = (const float*)d_in[24]; const float* fc2_b = (const float*)d_in[25];
  const float* fc3_w = (const float*)d_in[26]; const float* fc3_b = (const float*)d_in[27];

  float* outf = (float*)d_out;
  float* wsf  = (float*)d_ws;

  // workspace layout (floats)
  const size_t MISC = 0;
  const size_t H1   = 16;
  const size_t S_H1 = (size_t)64 * 32 * 126 * 126;
  const size_t H2   = H1 + S_H1;
  const size_t S_H2 = (size_t)64 * 32 * 62 * 62;
  const size_t PB   = H2 + S_H2;
  const size_t S_PB = (size_t)64 * 128 * 30 * 30;
  const size_t UB   = PB + S_PB;
  const size_t S_UB = (size_t)512 * 128 * 28 * 28;
  const size_t CB   = UB + S_UB;
  const size_t S_CB = (size_t)64 * 8 * 16 * 28 * 28;
  const size_t SEL  = CB + S_CB;
  const size_t R1   = SEL + 64 * 16;
  const size_t R2   = R1 + (size_t)64 * 512;

  init_misc<<<1, 32, 0, stream>>>(wsf + MISC);

  // conv1: 3->32, 128 -> 126, relu
  conv_wmma<<<dim3(64, (15876 + 31) / 32), 64, 0, stream>>>(
      x, conv1_w, conv1_b, wsf + H1, 3, 128, 128, 32, 3, 3, 1, 126, 126, 1);
  layernorm_kernel<<<64, 256, 0, stream>>>(wsf + H1, ln1_w, ln1_b, 32 * 126 * 126);

  // conv2: 32->32, stride 2 -> 62, relu
  conv_wmma<<<dim3(64, (3844 + 31) / 32), 64, 0, stream>>>(
      wsf + H1, conv2_w, conv2_b, wsf + H2, 32, 126, 126, 32, 3, 3, 2, 62, 62, 1);
  layernorm_kernel<<<64, 256, 0, stream>>>(wsf + H2, ln2_w, ln2_b, 32 * 62 * 62);

  // primary caps conv: 32->128, stride 2 -> 30; then squash over 16-dim
  conv_wmma<<<dim3(64, (900 + 31) / 32), 256, 0, stream>>>(
      wsf + H2, pc_w, pc_b, wsf + PB, 32, 62, 62, 128, 3, 3, 2, 30, 30, 0);
  squash_primary<<<(512 * 900 + 255) / 256, 256, 0, stream>>>(wsf + PB, 512, 900);

  // caps1: (512,16,30,30) -> votes (512,128,28,28); route -> (64,8,16,784)
  conv_wmma<<<dim3(512, (784 + 31) / 32), 256, 0, stream>>>(
      wsf + PB, w1, b1, wsf + UB, 16, 30, 30, 128, 3, 3, 1, 28, 28, 0);
  routing_kernel<8><<<(64 * 784 + 7) / 8, 256, 0, stream>>>(wsf + UB, wsf + CB, wsf + MISC + 0, 64, 784);

  // caps2: stride 2 -> 13
  conv_wmma<<<dim3(512, (169 + 31) / 32), 256, 0, stream>>>(
      wsf + CB, w2, b2, wsf + UB, 16, 28, 28, 128, 3, 3, 2, 13, 13, 0);
  routing_kernel<8><<<(64 * 169 + 7) / 8, 256, 0, stream>>>(wsf + UB, wsf + CB, wsf + MISC + 1, 64, 169);

  // caps3: 5x5 -> 9
  conv_wmma<<<dim3(512, (81 + 31) / 32), 256, 0, stream>>>(
      wsf + CB, w3, b3, wsf + UB, 16, 13, 13, 128, 5, 5, 1, 9, 9, 0);
  routing_kernel<8><<<(64 * 81 + 7) / 8, 256, 0, stream>>>(wsf + UB, wsf + CB, wsf + MISC + 2, 64, 81);

  // caps4: stride 2 -> 4
  conv_wmma<<<dim3(512, 1), 256, 0, stream>>>(
      wsf + CB, w4, b4, wsf + UB, 16, 9, 9, 128, 3, 3, 2, 4, 4, 0);
  routing_kernel<8><<<(64 * 16 + 7) / 8, 256, 0, stream>>>(wsf + UB, wsf + CB, wsf + MISC + 3, 64, 16);

  // caps5: 4x4 -> 1, Cout=32 (2 out_caps x 16)
  conv_wmma<<<dim3(512, 1), 64, 0, stream>>>(
      wsf + CB, w5, b5, wsf + UB, 16, 4, 4, 32, 4, 4, 1, 1, 1, 0);
  routing_kernel<2><<<8, 256, 0, stream>>>(wsf + UB, wsf + CB, wsf + MISC + 4, 64, 1);

  // decoder: mask-select, fc1 relu, fc2 relu, fc3 sigmoid (into d_out+129)
  sel_kernel<<<4, 256, 0, stream>>>(wsf + CB, target, wsf + SEL);
  gemm_wmma<<<dim3(2, 4),   256, 0, stream>>>(wsf + SEL, fc1_w, fc1_b, wsf + R1, 64, 16,   512,   1);
  gemm_wmma<<<dim3(2, 8),   256, 0, stream>>>(wsf + R1,  fc2_w, fc2_b, wsf + R2, 64, 512,  1024,  1);
  gemm_wmma<<<dim3(2, 384), 256, 0, stream>>>(wsf + R2,  fc3_w, fc3_b, outf + 129, 64, 1024, 49152, 2);

  // outputs: class predictions [0,128), entropy [128], reconstruction [129,...)
  class_pred<<<1, 64, 0, stream>>>(wsf + CB, outf);
  ent_finalize<<<1, 1, 0, stream>>>(wsf + MISC, outf + 128);
}
